// Block_15109694947416
// MI455X (gfx1250) — matrix operand order repaired and checked by reading, not compiled, not run on hardware
//
#include <hip/hip_runtime.h>

// ---------------------------------------------------------------------------
// RWKV v4 block for MI455X (gfx1250, wave32, WMMA).
// GEMMs: v_wmma_f32_16x16x32_f16, double-buffered LDS fed by CDNA5
// GLOBAL_LOAD_ASYNC_TO_LDS_B128 (ASYNCcnt), pre-transposed f16 weights.
// WKV scan in fp32 (numerically-stable reference form).
// ---------------------------------------------------------------------------

typedef __attribute__((ext_vector_type(16))) _Float16 v16h;
typedef __attribute__((ext_vector_type(8)))  _Float16 v8h;
typedef __attribute__((ext_vector_type(8)))  float    v8f;
typedef __attribute__((__vector_size__(4 * sizeof(int)))) int v4i_t;

#define DEVI __device__ __forceinline__

constexpr int kB   = 8;
constexpr int kT   = 2048;
constexpr int kC   = 1024;
constexpr int kM   = kB * kT;     // 16384 tokens
constexpr int kC4  = 4 * kC;      // 4096
constexpr float kEPS = 1e-5f;

// epilogue modes for the WMMA GEMM
enum { EPI_NONE = 0, EPI_RESID, EPI_SIGMOID, EPI_SIGMUL_H, EPI_RELU2_H, EPI_RR_RESID };

DEVI v16h cat8(v8h lo, v8h hi) {
  return __builtin_shufflevector(lo, hi, 0, 1, 2, 3, 4, 5, 6, 7,
                                         8, 9, 10, 11, 12, 13, 14, 15);
}

DEVI float sigmoidf_fast(float x) { return 1.f / (1.f + __expf(-x)); }

// ---- CDNA5 async global->LDS copy (no VGPR data, tracked by ASYNCcnt) -----
#if __has_builtin(__builtin_amdgcn_global_load_async_to_lds_b128)
#define HAVE_ASYNC_LDS 1
#else
#define HAVE_ASYNC_LDS 0
#endif

#if HAVE_ASYNC_LDS
DEVI void async_cp_b128(const _Float16* g, _Float16* l) {
  __builtin_amdgcn_global_load_async_to_lds_b128(
      (__attribute__((address_space(1))) v4i_t*)(unsigned long long)(g),
      (__attribute__((address_space(3))) v4i_t*)(l),
      0, 0);
}
#endif

DEVI void wait_async0() {
#if __has_builtin(__builtin_amdgcn_s_wait_asynccnt)
  __builtin_amdgcn_s_wait_asynccnt(0);
#else
  asm volatile("s_wait_asynccnt 0x0" ::: "memory");
#endif
}

// ---------------------------------------------------------------------------
// fp32 (Kd x Nd, row-major) -> fp16 transposed (Nd x Kd, row-major)
// Kd, Nd are powers of two; pass log2 shifts to avoid integer division.
// ---------------------------------------------------------------------------
__global__ __launch_bounds__(256) void cvt_t_kernel(const float* __restrict__ s,
                                                    _Float16* __restrict__ d,
                                                    unsigned total, int kshift, int nshift) {
  const unsigned i = blockIdx.x * 256u + threadIdx.x;   // index into dst (n*Kd + k)
  if (i >= total) return;
  const unsigned k = i & ((1u << kshift) - 1u);
  const unsigned n = i >> kshift;
  d[i] = (_Float16)s[(k << nshift) + n];
}

// ---------------------------------------------------------------------------
// LayerNorm over C=1024, one block (256 threads) per token
// ---------------------------------------------------------------------------
__global__ __launch_bounds__(256) void ln_kernel(const float* __restrict__ x,
                                                 const float* __restrict__ gw,
                                                 const float* __restrict__ bw,
                                                 float* __restrict__ out) {
  __shared__ float red[256];
  const int tid = threadIdx.x;
  const unsigned base = blockIdx.x * (unsigned)kC;

  float v0 = x[base + tid];
  float v1 = x[base + tid + 256];
  float v2 = x[base + tid + 512];
  float v3 = x[base + tid + 768];

  red[tid] = v0 + v1 + v2 + v3;
  __syncthreads();
  for (int st = 128; st > 0; st >>= 1) {
    if (tid < st) red[tid] += red[tid + st];
    __syncthreads();
  }
  const float mean = red[0] * (1.f / (float)kC);
  __syncthreads();

  const float d0 = v0 - mean, d1 = v1 - mean, d2 = v2 - mean, d3 = v3 - mean;
  red[tid] = d0 * d0 + d1 * d1 + d2 * d2 + d3 * d3;
  __syncthreads();
  for (int st = 128; st > 0; st >>= 1) {
    if (tid < st) red[tid] += red[tid + st];
    __syncthreads();
  }
  const float rstd = rsqrtf(red[0] * (1.f / (float)kC) + kEPS);

  out[base + tid]       = d0 * rstd * gw[tid]       + bw[tid];
  out[base + tid + 256] = d1 * rstd * gw[tid + 256] + bw[tid + 256];
  out[base + tid + 512] = d2 * rstd * gw[tid + 512] + bw[tid + 512];
  out[base + tid + 768] = d3 * rstd * gw[tid + 768] + bw[tid + 768];
}

// ---------------------------------------------------------------------------
// time-shift mixes (attention: 3 outputs, FFN: 2 outputs), f16 results
// ---------------------------------------------------------------------------
__global__ __launch_bounds__(256) void mix3_kernel(const float* __restrict__ h,
                                                   const float* __restrict__ tk,
                                                   const float* __restrict__ tv,
                                                   const float* __restrict__ tr,
                                                   _Float16* __restrict__ xk,
                                                   _Float16* __restrict__ xv,
                                                   _Float16* __restrict__ xr) {
  const unsigned i = blockIdx.x * 256u + threadIdx.x;
  if (i >= (unsigned)kM * kC) return;
  const int c = (int)(i & (kC - 1));
  const int t = (int)((i >> 10) & (kT - 1));
  const float hv = h[i];
  const float hp = (t == 0) ? 0.f : h[i - kC];
  const float ak = tk[c], av = tv[c], ar = tr[c];
  xk[i] = (_Float16)(hv * ak + hp * (1.f - ak));
  xv[i] = (_Float16)(hv * av + hp * (1.f - av));
  xr[i] = (_Float16)(hv * ar + hp * (1.f - ar));
}

__global__ __launch_bounds__(256) void mix2_kernel(const float* __restrict__ g,
                                                   const float* __restrict__ tk,
                                                   const float* __restrict__ tr,
                                                   _Float16* __restrict__ gk,
                                                   _Float16* __restrict__ gr) {
  const unsigned i = blockIdx.x * 256u + threadIdx.x;
  if (i >= (unsigned)kM * kC) return;
  const int c = (int)(i & (kC - 1));
  const int t = (int)((i >> 10) & (kT - 1));
  const float gv = g[i];
  const float gp = (t == 0) ? 0.f : g[i - kC];
  const float ak = tk[c], ar = tr[c];
  gk[i] = (_Float16)(gv * ak + gp * (1.f - ak));
  gr[i] = (_Float16)(gv * ar + gp * (1.f - ar));
}

// ---------------------------------------------------------------------------
// Numerically-stable WKV scan, fp32, one thread per (b, c) channel
// ---------------------------------------------------------------------------
__global__ __launch_bounds__(256) void wkv_kernel(const float* __restrict__ td,
                                                  const float* __restrict__ tf,
                                                  const float* __restrict__ k,
                                                  const float* __restrict__ v,
                                                  float* __restrict__ y) {
  const int tid = blockIdx.x * 256 + threadIdx.x;   // 0 .. B*C-1
  if (tid >= kB * kC) return;
  const int b = tid >> 10;
  const int c = tid & (kC - 1);
  const float w = -__expf(td[c]);
  const float u = tf[c];

  float aa = 0.f, bb = 0.f, pp = -1e30f;
  unsigned idx = (unsigned)b * kT * kC + c;
  for (int t = 0; t < kT; ++t, idx += kC) {
    const float kt = k[idx];
    const float vt = v[idx];
    const float ww = u + kt;
    const float p  = fmaxf(pp, ww);
    const float e1 = __expf(pp - p);
    const float e2 = __expf(ww - p);
    y[idx] = (e1 * aa + e2 * vt) / (e1 * bb + e2);
    const float ww2 = pp + w;
    const float p2  = fmaxf(ww2, kt);
    const float f1  = __expf(ww2 - p2);
    const float f2  = __expf(kt - p2);
    aa = f1 * aa + f2 * vt;
    bb = f1 * bb + f2;
    pp = p2;
  }
}

// ---------------------------------------------------------------------------
// Tiled WMMA GEMM: out[MxN] = A[MxK] @ W[KxN], W supplied pre-transposed as
// Bt[NxK].  WG tile 64x256, 8 waves each 32x64 (2x4 WMMA subtiles), K-step 32,
// double-buffered LDS with async global->LDS staging (no staging VGPRs, no
// spills under the 128-VGPR budget), one barrier per K-step.
//
// Fragment layouts per CDNA5 ISA:
//  A (16x32 f16): lane m=L&15; halves 0..7 -> K kb..kb+7, 8..15 -> K kb+16..,
//                 kb = (L>=16)*8.
//  B (32x16 f16): lane n=L&15; same K order as A (kb..kb+7, kb+16..kb+23).
//  C/D (16x16 f32, v8f): element r -> M=(L>=16)*8+r, N=L&15.
// ---------------------------------------------------------------------------
template <int EPI>
__global__
__attribute__((amdgpu_flat_work_group_size(256, 256)))
void gemm_wmma_kernel(
    const _Float16* __restrict__ A,   // M x K row-major (f16 activations)
    const _Float16* __restrict__ Bt,  // N x K row-major (f16 weights, transposed)
    float* __restrict__ outF, _Float16* __restrict__ outH,
    const float* __restrict__ resid, const float* __restrict__ scale,
    int M, int N, int K) {
  constexpr int LDST = 40;  // halves; 80B rows keep every sub-chunk 16B-aligned
  __shared__ _Float16 As[2][64 * LDST];
  __shared__ _Float16 Bs[2][256 * LDST];

  const int tid  = threadIdx.x;
  const int lane = tid & 31;
  const int wave = tid >> 5;
  const int wr = wave >> 2;                  // 0..1 : 32-row strip (M)
  const int wc = wave & 3;                   // 0..3 : 64-col strip (N)
  const unsigned m_wg = blockIdx.y * 64u;
  const unsigned n_wg = blockIdx.x * 256u;

  // staging coordinates: one 16B chunk per load
  const int srow = tid >> 2;                 // 0..63
  const int scg  = (tid & 3) * 8;            // 0/8/16/24
  const unsigned aoff = (m_wg + srow) * (unsigned)K + scg;
  unsigned boff[4];
#pragma unroll
  for (int j = 0; j < 4; ++j)
    boff[j] = (n_wg + srow + 64u * j) * (unsigned)K + scg;
  const int aSt = srow * LDST + scg;         // LDS staging indices (halves)
  int bSt[4];
#pragma unroll
  for (int j = 0; j < 4; ++j) bSt[j] = (srow + 64 * j) * LDST + scg;

  // fragment LDS base indices (in halves)
  const int lrow = lane & 15;
  const int kbA  = (lane >> 4) * 8;
  const int kbB  = (lane >> 4) * 8;
  const int aIdx[2] = { (wr * 32 + lrow) * LDST + kbA,
                        (wr * 32 + 16 + lrow) * LDST + kbA };
  int bIdx[4];
#pragma unroll
  for (int ni = 0; ni < 4; ++ni)
    bIdx[ni] = (wc * 64 + ni * 16 + lrow) * LDST + kbB;

  v8f acc[2][4] = {};

#if HAVE_ASYNC_LDS
  // ---- async double-buffered pipeline: zero staging VGPRs ----
  async_cp_b128(A + aoff, &As[0][aSt]);
#pragma unroll
  for (int j = 0; j < 4; ++j) async_cp_b128(Bt + boff[j], &Bs[0][bSt[j]]);
  wait_async0();
  __syncthreads();

  int buf = 0;
  for (unsigned k0 = 0; k0 < (unsigned)K; k0 += 32) {
    const bool more = (k0 + 32) < (unsigned)K;
    if (more) {   // async-stage next tile into the other buffer
      const int nb = buf ^ 1;
      async_cp_b128(A + aoff + k0 + 32, &As[nb][aSt]);
#pragma unroll
      for (int j = 0; j < 4; ++j)
        async_cp_b128(Bt + boff[j] + k0 + 32, &Bs[nb][bSt[j]]);
    }

    const _Float16* as = As[buf];
    const _Float16* bs = Bs[buf];
    v16h af[2], bf[4];
#pragma unroll
    for (int mi = 0; mi < 2; ++mi)
      af[mi] = cat8(*(const v8h*)(as + aIdx[mi]),
                    *(const v8h*)(as + aIdx[mi] + 16));
#pragma unroll
    for (int ni = 0; ni < 4; ++ni)
      bf[ni] = cat8(*(const v8h*)(bs + bIdx[ni]),
                    *(const v8h*)(bs + bIdx[ni] + 16));

#pragma unroll
    for (int mi = 0; mi < 2; ++mi)
#pragma unroll
      for (int ni = 0; ni < 4; ++ni)
        acc[mi][ni] = __builtin_amdgcn_wmma_f32_16x16x32_f16(
            false, af[mi], false, bf[ni], (short)0, acc[mi][ni], false, false);

    if (more) {
      wait_async0();      // this wave's async copies into nb are done
      __syncthreads();    // everyone's are done
      buf ^= 1;
    }
  }
#else
  // ---- fallback: register-prefetch double buffering ----
  {
    const uint4 a4 = *(const uint4*)(A + aoff);
    uint4 b4[4];
#pragma unroll
    for (int j = 0; j < 4; ++j) b4[j] = *(const uint4*)(Bt + boff[j]);
    *(uint4*)(&As[0][aSt]) = a4;
#pragma unroll
    for (int j = 0; j < 4; ++j) *(uint4*)(&Bs[0][bSt[j]]) = b4[j];
  }
  __syncthreads();

  int buf = 0;
  for (unsigned k0 = 0; k0 < (unsigned)K; k0 += 32) {
    const bool more = (k0 + 32) < (unsigned)K;
    uint4 a4, b4[4];
    if (more) {
      a4 = *(const uint4*)(A + aoff + k0 + 32);
#pragma unroll
      for (int j = 0; j < 4; ++j) b4[j] = *(const uint4*)(Bt + boff[j] + k0 + 32);
    }

    const _Float16* as = As[buf];
    const _Float16* bs = Bs[buf];
    v16h af[2], bf[4];
#pragma unroll
    for (int mi = 0; mi < 2; ++mi)
      af[mi] = cat8(*(const v8h*)(as + aIdx[mi]),
                    *(const v8h*)(as + aIdx[mi] + 16));
#pragma unroll
    for (int ni = 0; ni < 4; ++ni)
      bf[ni] = cat8(*(const v8h*)(bs + bIdx[ni]),
                    *(const v8h*)(bs + bIdx[ni] + 16));

#pragma unroll
    for (int mi = 0; mi < 2; ++mi)
#pragma unroll
      for (int ni = 0; ni < 4; ++ni)
        acc[mi][ni] = __builtin_amdgcn_wmma_f32_16x16x32_f16(
            false, af[mi], false, bf[ni], (short)0, acc[mi][ni], false, false);

    if (more) {
      const int nb = buf ^ 1;
      *(uint4*)(&As[nb][aSt]) = a4;
#pragma unroll
      for (int j = 0; j < 4; ++j) *(uint4*)(&Bs[nb][bSt[j]]) = b4[j];
      __syncthreads();
      buf = nb;
    }
  }
#endif

  // epilogue: 32-bit offsets only (all tensors < 2^31 elements)
  const int rbase = (lane >> 4) * 8;
#pragma unroll
  for (int mi = 0; mi < 2; ++mi)
#pragma unroll
    for (int ni = 0; ni < 4; ++ni) {
      const unsigned m0 = m_wg + wr * 32 + mi * 16 + rbase;
      const unsigned n  = n_wg + wc * 64 + ni * 16 + lrow;
      unsigned o = m0 * (unsigned)N + n;
#pragma unroll
      for (int r = 0; r < 8; ++r, o += (unsigned)N) {
        const float a = acc[mi][ni][r];
        if constexpr (EPI == EPI_NONE) {
          outF[o] = a;
        } else if constexpr (EPI == EPI_RESID) {
          outF[o] = a + resid[o];
        } else if constexpr (EPI == EPI_SIGMOID) {
          outF[o] = sigmoidf_fast(a);
        } else if constexpr (EPI == EPI_SIGMUL_H) {
          outH[o] = (_Float16)(sigmoidf_fast(a) * scale[o]);
        } else if constexpr (EPI == EPI_RELU2_H) {
          const float t = a > 0.f ? a : 0.f;
          outH[o] = (_Float16)(t * t);
        } else {  // EPI_RR_RESID
          outF[o] = resid[o] + scale[o] * a;
        }
      }
    }
}

// ---------------------------------------------------------------------------
// host launcher
// ---------------------------------------------------------------------------
extern "C" void kernel_launch(void* const* d_in, const int* in_sizes, int n_in,
                              void* d_out, int out_size, void* d_ws, size_t ws_size,
                              hipStream_t stream) {
  const float* x    = (const float*)d_in[0];
  const float* ln1g = (const float*)d_in[1];
  const float* ln1b = (const float*)d_in[2];
  const float* ln2g = (const float*)d_in[3];
  const float* ln2b = (const float*)d_in[4];
  const float* tdec = (const float*)d_in[5];
  const float* tfir = (const float*)d_in[6];
  const float* tmk  = (const float*)d_in[7];
  const float* tmv  = (const float*)d_in[8];
  const float* tmr  = (const float*)d_in[9];
  const float* Wk   = (const float*)d_in[10];
  const float* Wv   = (const float*)d_in[11];
  const float* Wr   = (const float*)d_in[12];
  const float* Wo   = (const float*)d_in[13];
  const float* ftmk = (const float*)d_in[14];
  const float* ftmr = (const float*)d_in[15];
  const float* fWk  = (const float*)d_in[16];
  const float* fWr  = (const float*)d_in[17];
  const float* fWv  = (const float*)d_in[18];
  (void)in_sizes; (void)n_in; (void)out_size; (void)ws_size;

  // ---- workspace layout (256B-aligned regions, explicit lifetime reuse) ----
  char* ws = (char*)d_ws;
  size_t off = 0;
  auto alloc = [&](size_t bytes) -> void* {
    void* p = ws + off;
    off += (bytes + 255) & ~(size_t)255;
    return p;
  };
  const size_t CC  = (size_t)kC * kC;
  const size_t MC  = (size_t)kM * kC;
  const size_t MC4 = (size_t)kM * kC4;

  // transposed f16 weights (N x K layouts)
  _Float16* Wk16  = (_Float16*)alloc(CC * 2);
  _Float16* Wv16  = (_Float16*)alloc(CC * 2);
  _Float16* Wr16  = (_Float16*)alloc(CC * 2);
  _Float16* Wo16  = (_Float16*)alloc(CC * 2);
  _Float16* fWk16 = (_Float16*)alloc(CC * 4 * 2);
  _Float16* fWr16 = (_Float16*)alloc(CC * 2);
  _Float16* fWv16 = (_Float16*)alloc(CC * 4 * 2);

  float*    hbuf = (float*)alloc(MC * 4);      // h  -> then y -> then rr
  _Float16* xk   = (_Float16*)alloc(MC * 2);   // xk -> then a16 (= sig(r)*y)
  _Float16* xv   = (_Float16*)alloc(MC * 2);   // xv -> then gk
  _Float16* xr   = (_Float16*)alloc(MC * 2);   // xr -> then gr
  float*    kbuf = (float*)alloc(MC * 4);      // k  -> then x1
  float*    vbuf = (float*)alloc(MC * 4);      // v  -> then g
  _Float16* kk16 = (_Float16*)alloc(MC4 * 2);  // relu^2(gk @ fWk) in f16

  float*    ybuf  = hbuf;
  float*    rrbuf = hbuf;
  _Float16* a16   = xk;
  _Float16* gk    = xv;
  _Float16* gr    = xr;
  float*    x1    = kbuf;
  float*    gbuf  = vbuf;
  float*    out   = (float*)d_out;

  // ---- convert + transpose weights (one-time, ~26MB) ----
  auto cvtT = [&](const float* s, _Float16* d, unsigned total, int ksh, int nsh) {
    cvt_t_kernel<<<(total + 255) / 256, 256, 0, stream>>>(s, d, total, ksh, nsh);
  };
  cvtT(Wk,  Wk16,  (unsigned)CC,      10, 10);  // 1024x1024 -> 1024x1024
  cvtT(Wv,  Wv16,  (unsigned)CC,      10, 10);
  cvtT(Wr,  Wr16,  (unsigned)CC,      10, 10);
  cvtT(Wo,  Wo16,  (unsigned)CC,      10, 10);
  cvtT(fWk, fWk16, (unsigned)(CC*4),  10, 12);  // K=1024, N=4096 -> 4096x1024
  cvtT(fWr, fWr16, (unsigned)CC,      10, 10);
  cvtT(fWv, fWv16, (unsigned)(CC*4),  12, 10);  // K=4096, N=1024 -> 1024x4096

  const unsigned ewGrid = (unsigned)(MC / 256);  // 65536
  const dim3 gC (kC  / 256, kM / 64);            // (4, 256)
  const dim3 gC4(kC4 / 256, kM / 64);            // (16, 256)

  // ---- time-mix (attention) branch ----
  ln_kernel<<<kM, 256, 0, stream>>>(x, ln1g, ln1b, hbuf);
  mix3_kernel<<<ewGrid, 256, 0, stream>>>(hbuf, tmk, tmv, tmr, xk, xv, xr);

  gemm_wmma_kernel<EPI_NONE><<<gC, 256, 0, stream>>>(
      xk, Wk16, kbuf, nullptr, nullptr, nullptr, kM, kC, kC);
  gemm_wmma_kernel<EPI_NONE><<<gC, 256, 0, stream>>>(
      xv, Wv16, vbuf, nullptr, nullptr, nullptr, kM, kC, kC);

  wkv_kernel<<<(kB * kC) / 256, 256, 0, stream>>>(tdec, tfir, kbuf, vbuf, ybuf);

  // a16 = f16( sigmoid(xr @ Wr) * y )   (fused epilogue)
  gemm_wmma_kernel<EPI_SIGMUL_H><<<gC, 256, 0, stream>>>(
      xr, Wr16, nullptr, a16, nullptr, ybuf, kM, kC, kC);

  // x1 = x + a16 @ Wo
  gemm_wmma_kernel<EPI_RESID><<<gC, 256, 0, stream>>>(
      a16, Wo16, x1, nullptr, x, nullptr, kM, kC, kC);

  // ---- channel-mix (FFN) branch ----
  ln_kernel<<<kM, 256, 0, stream>>>(x1, ln2g, ln2b, gbuf);
  mix2_kernel<<<ewGrid, 256, 0, stream>>>(gbuf, ftmk, ftmr, gk, gr);

  // kk = f16( relu(gk @ fWk)^2 )
  gemm_wmma_kernel<EPI_RELU2_H><<<gC4, 256, 0, stream>>>(
      gk, fWk16, nullptr, kk16, nullptr, nullptr, kM, kC4, kC);

  // rr = sigmoid(gr @ fWr)
  gemm_wmma_kernel<EPI_SIGMOID><<<gC, 256, 0, stream>>>(
      gr, fWr16, rrbuf, nullptr, nullptr, nullptr, kM, kC, kC);

  // out = x1 + rr * (kk @ fWv)
  gemm_wmma_kernel<EPI_RR_RESID><<<gC, 256, 0, stream>>>(
      kk16, fWv16, out, nullptr, x1, rrbuf, kM, kC, kC4);
}
